// RegionSelector_69741678953208
// MI455X (gfx1250) — matrix operand order repaired and checked by reading, not compiled, not run on hardware
//
#include <hip/hip_runtime.h>

// ---------------------------------------------------------------------------
// MI455X / gfx1250. wave32, WMMA 16x16x32 bf16 (f32 accum).
// All GEMM operands pre-converted to bf16; weights pre-transposed to [N,K]
// so both A and B fragments are contiguous bf16 vector loads (no VALU
// conversion in the hot loop).
// Shapes: B=16, N=196, D=1024, H=8, DK=128, M=16384, MB=1024, TOPK=32, C=29.
// ---------------------------------------------------------------------------

typedef __attribute__((ext_vector_type(16))) __bf16 v16bf;
typedef __attribute__((ext_vector_type(8)))  __bf16 v8bf;
typedef __attribute__((ext_vector_type(8)))  float  v8f;

#define B_   16
#define N_   196
#define D_   1024
#define H_   8
#define DK_  128
#define MB_  1024
#define C_   29
#define NEG_INF (-3.0e38f)

__device__ __forceinline__ unsigned short bf16bits(float f) {
  // round-to-nearest-even f32 -> bf16 bit pattern
  unsigned int u = __builtin_bit_cast(unsigned int, f);
  return (unsigned short)((u + 0x7FFFu + ((u >> 16) & 1u)) >> 16);
}

__device__ __forceinline__ float gelu_tanh(float x) {
  const float c = 0.7978845608028654f;
  float t = tanhf(c * (x + 0.044715f * x * x * x));
  return 0.5f * x * (1.0f + t);
}

// A fragment (16x32 bf16, MxK): lane laneLo owns row,
// element s holds K = hi*16 + s (contiguous, matching B).
// Two contiguous 16-byte bf16 loads.
__device__ __forceinline__ v16bf load_a_bf(const unsigned short* __restrict__ A,
                                           int row, int lda, int k0, int hi) {
  const __bf16* base = (const __bf16*)A + (size_t)row * lda + k0 + hi * 16;
  v8bf lo = *(const v8bf*)base;
  v8bf hh = *(const v8bf*)(base + 8);
  return __builtin_shufflevector(lo, hh, 0, 1, 2, 3, 4, 5, 6, 7,
                                 8, 9, 10, 11, 12, 13, 14, 15);
}

// B fragment (32x16 bf16, KxN): lane laneLo owns col; lanes 0-15 hold
// K={0..15}, lanes 16-31 hold K={16..31}.  With W pre-transposed to [N,K]
// this is 16 *contiguous* bf16 = one 32-byte aligned load.
__device__ __forceinline__ v16bf load_b_bf(const unsigned short* __restrict__ Wt,
                                           int col, int ldk, int k0, int hi) {
  return *(const v16bf*)((const __bf16*)Wt + (size_t)col * ldk + k0 + hi * 16);
}

// ---------------------------------------------------------------------------
// elementwise f32 -> bf16 (float4 -> ushort4), counts are multiples of 4
// ---------------------------------------------------------------------------
__global__ void cvt_bf16(const float* __restrict__ in,
                         unsigned short* __restrict__ out, int n4) {
  const int i = blockIdx.x * 256 + threadIdx.x;
  if (i >= n4) return;
  const float4 v = ((const float4*)in)[i];
  ushort4 o;
  o.x = bf16bits(v.x); o.y = bf16bits(v.y);
  o.z = bf16bits(v.z); o.w = bf16bits(v.w);
  ((ushort4*)out)[i] = o;
}

// ---------------------------------------------------------------------------
// W[K,N] f32 -> Wt[N,K] bf16, LDS-tiled 32x32 transpose (coalesced both ways)
// ---------------------------------------------------------------------------
__global__ __launch_bounds__(256) void transpose_cvt(
    const float* __restrict__ W, unsigned short* __restrict__ Wt,
    int K, int N) {
  __shared__ unsigned short tile[32][33];
  const int k0 = blockIdx.x * 32, n0 = blockIdx.y * 32;
  const int tx = threadIdx.x & 31, ty = threadIdx.x >> 5;  // 8 rows / pass
#pragma unroll
  for (int r = ty; r < 32; r += 8)
    tile[r][tx] = bf16bits(W[(size_t)(k0 + r) * N + n0 + tx]);
  __syncthreads();
#pragma unroll
  for (int r = ty; r < 32; r += 8)
    Wt[(size_t)(n0 + r) * K + k0 + tx] = tile[tx][r];
}

// ---------------------------------------------------------------------------
// C[M,N] = act(A[M,K] @ Wt[N,K]^T + bias).  Block tile 128x128, 8 waves
// (4x2), each wave 32x64 = 2x4 WMMA subtiles, K step 32.  A, Wt bf16;
// output f32 or bf16.
// ---------------------------------------------------------------------------
__global__ __launch_bounds__(256) void gemm_bf16(
    const unsigned short* __restrict__ A, const unsigned short* __restrict__ Wt,
    const float* __restrict__ bias, void* __restrict__ Cout,
    int M, int K, int N, int act, int out_bf16) {
  const int tid = threadIdx.x;
  const int lane = tid & 31, laneLo = lane & 15, hi = lane >> 4;
  const int wid = tid >> 5;
  const int m0 = blockIdx.x * 128 + (wid & 3) * 32;
  const int n0 = blockIdx.y * 128 + (wid >> 2) * 64;

  const int rowA0 = min(m0 + laneLo, M - 1);
  const int rowA1 = min(m0 + 16 + laneLo, M - 1);

  v8f acc[2][4] = {};
  for (int k0 = 0; k0 < K; k0 += 32) {
    if (k0 + 32 < K) {  // L2-resident streams; hint the next K tile
      __builtin_prefetch((const __bf16*)A + (size_t)rowA0 * K + k0 + 32, 0, 1);
      __builtin_prefetch(
          (const __bf16*)Wt + (size_t)(n0 + laneLo) * K + k0 + 32, 0, 1);
    }
    const v16bf a0 = load_a_bf(A, rowA0, K, k0, hi);
    const v16bf a1 = load_a_bf(A, rowA1, K, k0, hi);
#pragma unroll
    for (int j = 0; j < 4; ++j) {
      const v16bf bf = load_b_bf(Wt, n0 + 16 * j + laneLo, K, k0, hi);
      acc[0][j] = __builtin_amdgcn_wmma_f32_16x16x32_bf16(
          false, a0, false, bf, (short)0, acc[0][j], false, false);
      acc[1][j] = __builtin_amdgcn_wmma_f32_16x16x32_bf16(
          false, a1, false, bf, (short)0, acc[1][j], false, false);
    }
  }
#pragma unroll
  for (int i = 0; i < 2; ++i)
#pragma unroll
    for (int j = 0; j < 4; ++j) {
      const int col = n0 + 16 * j + laneLo;
      const float bcol = bias[col];
#pragma unroll
      for (int r = 0; r < 8; ++r) {
        const int row = m0 + 16 * i + 8 * hi + r;
        if (row < M) {
          float v = acc[i][j][r] + bcol;
          if (act) v = gelu_tanh(v);
          if (out_bf16)
            ((unsigned short*)Cout)[(size_t)row * N + col] = bf16bits(v);
          else
            ((float*)Cout)[(size_t)row * N + col] = v;
        }
      }
    }
}

// ---------------------------------------------------------------------------
// scores[bh, n, m] = (q[b,n,h,:] . k[b,m,h,:]) / sqrt(DK).  q,k bf16 with
// contiguous K per lane; one wave per (bh, 16-row, 64-col) tile; 4 WMMA steps.
// ---------------------------------------------------------------------------
__global__ __launch_bounds__(256) void attn_scores(
    const unsigned short* __restrict__ Q, const unsigned short* __restrict__ Kb,
    float* __restrict__ S) {
  const int tid = threadIdx.x;
  const int lane = tid & 31, laneLo = lane & 15, hi = lane >> 4;
  const int wid = blockIdx.x * 8 + (tid >> 5);
  const int NT = 16, MT = 13;  // 13*16=208 rows cover 196, 16*64=1024 cols
  const int nt = wid % NT;
  const int mt = (wid / NT) % MT;
  const int bh = wid / (NT * MT);
  const int b = bh >> 3, h = bh & 7;
  const int m0 = mt * 16, n0 = nt * 64;
  const int qrow = b * N_ + min(m0 + laneLo, N_ - 1);

  v8f acc[4] = {};
#pragma unroll
  for (int k0 = 0; k0 < DK_; k0 += 32) {
    const v16bf a = load_a_bf(Q, qrow, D_, h * DK_ + k0, hi);
#pragma unroll
    for (int j = 0; j < 4; ++j) {
      const int col = n0 + 16 * j + laneLo;
      const v16bf bf = *(const v16bf*)((const __bf16*)Kb +
          ((size_t)(b * MB_ + col)) * D_ + h * DK_ + k0 + hi * 16);
      acc[j] = __builtin_amdgcn_wmma_f32_16x16x32_bf16(
          false, a, false, bf, (short)0, acc[j], false, false);
    }
  }
  const float scale = 0.088388347648318447f;  // 1/sqrt(128)
#pragma unroll
  for (int j = 0; j < 4; ++j) {
    const int col = n0 + 16 * j + laneLo;
#pragma unroll
    for (int r = 0; r < 8; ++r) {
      const int m = m0 + 8 * hi + r;
      if (m < N_) S[((size_t)bh * N_ + m) * MB_ + col] = acc[j][r] * scale;
    }
  }
}

// ---------------------------------------------------------------------------
// One wave per (b,h,n) row: top-32 of 1024 via 32 iterative wave-argmax
// passes, softmax over selected, weighted gather of v rows -> att (bf16).
// ---------------------------------------------------------------------------
__global__ __launch_bounds__(256) void topk_softmax_gather(
    const float* __restrict__ S, const float* __restrict__ V,
    unsigned short* __restrict__ Att) {
  const int lane = threadIdx.x & 31;
  const int gid = blockIdx.x * 8 + (threadIdx.x >> 5);  // < B*H*N exactly
  const int bh = gid / N_;
  const int n = gid - bh * N_;
  const int b = bh >> 3, h = bh & 7;
  const float* srow = S + (size_t)gid * MB_;

  float val[32];
#pragma unroll
  for (int i = 0; i < 32; ++i) val[i] = srow[i * 32 + lane];

  float selv = NEG_INF;
  int seli = 0;
  for (int t = 0; t < 32; ++t) {
    float mv = val[0];
    int mi = lane;
#pragma unroll
    for (int i = 1; i < 32; ++i) {
      const int idx = i * 32 + lane;
      if (val[i] > mv) { mv = val[i]; mi = idx; }
    }
    for (int off = 16; off > 0; off >>= 1) {
      const float ov = __shfl_xor(mv, off, 32);
      const int oi = __shfl_xor(mi, off, 32);
      if (ov > mv || (ov == mv && oi < mi)) { mv = ov; mi = oi; }
    }
    const bool mine = (mi & 31) == lane;
    const int slot = mi >> 5;
#pragma unroll
    for (int i = 0; i < 32; ++i)
      val[i] = (mine && slot == i) ? NEG_INF : val[i];
    if (lane == t) { selv = mv; seli = mi; }
  }

  // softmax over the 32 selected scores (lane t holds t-th largest)
  const float vmax = __shfl(selv, 0, 32);
  float p = __expf(selv - vmax);
  float sum = p;
  for (int off = 16; off > 0; off >>= 1) sum += __shfl_xor(sum, off, 32);
  p /= sum;

  // att[:, dk] = sum_t p_t * v[idx_t, dk]; 128 dk / 32 lanes = float4/lane
  float4 acc = {0.f, 0.f, 0.f, 0.f};
  const float* vbase = V + ((size_t)b * MB_) * D_ + h * DK_ + lane * 4;
  for (int t = 0; t < 32; ++t) {
    const float pt = __shfl(p, t, 32);
    const int it = __shfl(seli, t, 32);
    const float4 vv = *(const float4*)(vbase + (size_t)it * D_);
    acc.x += pt * vv.x; acc.y += pt * vv.y;
    acc.z += pt * vv.z; acc.w += pt * vv.w;
  }
  ushort4 o;
  o.x = bf16bits(acc.x); o.y = bf16bits(acc.y);
  o.z = bf16bits(acc.z); o.w = bf16bits(acc.w);
  *(ushort4*)(Att + ((size_t)(b * N_ + n)) * D_ + h * DK_ + lane * 4) = o;
}

// concat([xg, responses], axis=-1) -> F[rows, 2048] (all bf16)
__global__ void concat2(const unsigned short* __restrict__ X,
                        const unsigned short* __restrict__ R,
                        unsigned short* __restrict__ F, int rows) {
  const int idx = blockIdx.x * 256 + threadIdx.x;
  if (idx >= rows * 2048) return;
  const int row = idx >> 11, col = idx & 2047;
  F[idx] = (col < 1024) ? X[(size_t)row * 1024 + col]
                        : R[(size_t)row * 1024 + col - 1024];
}

// logits = fused @ W_head + b_head; probs = sigmoid(logits). One wave/row.
__global__ __launch_bounds__(256) void head_sigmoid(
    const float* __restrict__ Fu, const float* __restrict__ Wh,
    const float* __restrict__ bh, float* __restrict__ logits,
    float* __restrict__ probs) {
  const int lane = threadIdx.x & 31;
  const int row = blockIdx.x * 8 + (threadIdx.x >> 5);  // < 3136 exactly
  const float* frow = Fu + (size_t)row * D_;
  float acc[C_];
#pragma unroll
  for (int c = 0; c < C_; ++c) acc[c] = 0.f;
  for (int k = lane; k < D_; k += 32) {
    const float f = frow[k];
    const float* wr = Wh + (size_t)k * C_;
#pragma unroll
    for (int c = 0; c < C_; ++c) acc[c] += f * wr[c];
  }
#pragma unroll
  for (int c = 0; c < C_; ++c)
    for (int off = 16; off > 0; off >>= 1) acc[c] += __shfl_xor(acc[c], off, 32);
  if (lane == 0) {
#pragma unroll
    for (int c = 0; c < C_; ++c) {
      const float l = acc[c] + bh[c];
      logits[(size_t)row * C_ + c] = l;
      probs[(size_t)row * C_ + c] = 1.0f / (1.0f + __expf(-l));
    }
  }
}

// box_masks[i] = region_probs.reshape(-1)[box_labels[i] + boxes[i,0]*N] > 0.5
__global__ void box_mask(const int* __restrict__ boxes,
                         const int* __restrict__ labels,
                         const float* __restrict__ probs_flat,
                         float* __restrict__ out) {
  const int i = blockIdx.x * 256 + threadIdx.x;
  if (i >= 512) return;
  const int fi = labels[i] + boxes[i * 5] * N_;
  out[i] = (probs_flat[fi] > 0.5f) ? 1.0f : 0.0f;
}

// ---------------------------------------------------------------------------
extern "C" void kernel_launch(void* const* d_in, const int* in_sizes, int n_in,
                              void* d_out, int out_size, void* d_ws,
                              size_t ws_size, hipStream_t stream) {
  const float* x       = (const float*)d_in[0];
  const int*   boxes   = (const int*)d_in[1];
  const int*   labels  = (const int*)d_in[2];
  const float* W_ff    = (const float*)d_in[3];
  const float* b_ff    = (const float*)d_in[4];
  const float* memory  = (const float*)d_in[5];
  const float* W_mp    = (const float*)d_in[6];
  const float* b_mp    = (const float*)d_in[7];
  const float* W_q     = (const float*)d_in[8];
  const float* b_q     = (const float*)d_in[9];
  const float* W_k     = (const float*)d_in[10];
  const float* b_k     = (const float*)d_in[11];
  const float* W_v     = (const float*)d_in[12];
  const float* b_v     = (const float*)d_in[13];
  const float* W_o     = (const float*)d_in[14];
  const float* b_o     = (const float*)d_in[15];
  const float* W_fuse  = (const float*)d_in[16];
  const float* b_fuse  = (const float*)d_in[17];
  const float* W_head  = (const float*)d_in[18];
  const float* b_head  = (const float*)d_in[19];

  const size_t ROWS = (size_t)B_ * N_;          // 3136
  const size_t SZ_X = ROWS * D_;                // 3,211,264
  const size_t SZ_M = (size_t)B_ * MB_ * D_;    // 16,777,216
  const size_t SZ_S = (size_t)B_ * H_ * N_ * MB_;
  const size_t SZ_W = (size_t)D_ * D_;          // 1,048,576

  // ---- bf16 (u16) region ----
  unsigned short* wsu   = (unsigned short*)d_ws;
  unsigned short* xb    = wsu;            // x as bf16
  unsigned short* memb  = xb + SZ_X;      // memory as bf16
  unsigned short* Wff_t = memb + SZ_M;    // weights transposed [N,K] bf16
  unsigned short* Wmp_t = Wff_t + SZ_W;
  unsigned short* Wq_t  = Wmp_t + SZ_W;
  unsigned short* Wk_t  = Wq_t + SZ_W;
  unsigned short* Wv_t  = Wk_t + SZ_W;
  unsigned short* Wo_t  = Wv_t + SZ_W;
  unsigned short* Wfu_t = Wo_t + SZ_W;    // [1024, 2048] bf16
  unsigned short* xg    = Wfu_t + 2 * SZ_W;
  unsigned short* memp  = xg + SZ_X;
  unsigned short* qb    = memp + SZ_M;
  unsigned short* kb    = qb + SZ_X;
  unsigned short* attb  = kb + SZ_M;
  unsigned short* respb = attb + SZ_X;
  unsigned short* fuA   = respb + SZ_X;   // [3136, 2048] bf16
  // ---- f32 region (16B aligned: preceding u16 bytes are multiple of 16) ----
  float* vb    = (float*)(fuA + ROWS * 2048);
  float* sc    = vb + SZ_M;
  float* fused = sc + SZ_S;

  float* out_logits = (float*)d_out;
  float* out_probs  = out_logits + ROWS * C_;
  float* out_masks  = out_probs + ROWS * C_;

  const dim3 blk(256);
  const dim3 gX(25, 8);    // ceil(3136/128) x 1024/128
  const dim3 gM(128, 8);   // 16384/128 x 1024/128

  // operand conversions (one-time, bandwidth-trivial)
  cvt_bf16<<<(int)(SZ_X / 4 / 256), blk, 0, stream>>>(x, xb, (int)(SZ_X / 4));
  cvt_bf16<<<(int)(SZ_M / 4 / 256), blk, 0, stream>>>(memory, memb, (int)(SZ_M / 4));
  transpose_cvt<<<dim3(32, 32), blk, 0, stream>>>(W_ff, Wff_t, D_, D_);
  transpose_cvt<<<dim3(32, 32), blk, 0, stream>>>(W_mp, Wmp_t, D_, D_);
  transpose_cvt<<<dim3(32, 32), blk, 0, stream>>>(W_q, Wq_t, D_, D_);
  transpose_cvt<<<dim3(32, 32), blk, 0, stream>>>(W_k, Wk_t, D_, D_);
  transpose_cvt<<<dim3(32, 32), blk, 0, stream>>>(W_v, Wv_t, D_, D_);
  transpose_cvt<<<dim3(32, 32), blk, 0, stream>>>(W_o, Wo_t, D_, D_);
  transpose_cvt<<<dim3(64, 32), blk, 0, stream>>>(W_fuse, Wfu_t, 2048, D_);

  // xg = gelu(x @ W_ff + b_ff)            [bf16 out]
  gemm_bf16<<<gX, blk, 0, stream>>>(xb, Wff_t, b_ff, xg, 3136, D_, D_, 1, 1);
  // mem = memory @ W_mp + b_mp            [bf16 out]
  gemm_bf16<<<gM, blk, 0, stream>>>(memb, Wmp_t, b_mp, memp, 16384, D_, D_, 0, 1);
  // q/k bf16, v f32 (gather consumes f32)
  gemm_bf16<<<gX, blk, 0, stream>>>(xg, Wq_t, b_q, qb, 3136, D_, D_, 0, 1);
  gemm_bf16<<<gM, blk, 0, stream>>>(memp, Wk_t, b_k, kb, 16384, D_, D_, 0, 1);
  gemm_bf16<<<gM, blk, 0, stream>>>(memp, Wv_t, b_v, vb, 16384, D_, D_, 0, 0);
  // scores: 128 bh * 13 mtiles * 16 ntiles = 26624 waves / 8 = 3328 blocks
  attn_scores<<<3328, blk, 0, stream>>>(qb, kb, sc);
  // topk + softmax + gather: 25088 rows / 8 waves = 3136 blocks -> att bf16
  topk_softmax_gather<<<3136, blk, 0, stream>>>(sc, vb, attb);
  // responses = att @ W_o + b_o           [bf16 out]
  gemm_bf16<<<gX, blk, 0, stream>>>(attb, Wo_t, b_o, respb, 3136, D_, D_, 0, 1);
  // fused = concat(xg, responses) @ W_fuse + b_fuse   [f32 out for head]
  concat2<<<(int)((ROWS * 2048 + 255) / 256), blk, 0, stream>>>(xg, respb, fuA, 3136);
  gemm_bf16<<<gX, blk, 0, stream>>>(fuA, Wfu_t, b_fuse, fused, 3136, 2048, D_, 0, 0);
  // head + sigmoid -> d_out
  head_sigmoid<<<392, blk, 0, stream>>>(fused, W_head, b_head, out_logits, out_probs);
  // box masks
  box_mask<<<2, blk, 0, stream>>>(boxes, labels, out_probs, out_masks);
}
